// BiMambaBlock_55465207661156
// MI455X (gfx1250) — compile-verified, matrix operand order checked by reading, not run on hardware
//
#include <hip/hip_runtime.h>

// ---------------------------------------------------------------------------
// BiMamba block for MI455X (gfx1250, wave32, WMMA + async-to-LDS staging).
// All matmuls run through v_wmma_f32_16x16x32_bf16 with f32 accumulation.
// GEMM panels are staged into LDS with global_load_async_to_lds_b128
// (ASYNCcnt) and double-buffered across K-steps.
// ---------------------------------------------------------------------------

typedef __attribute__((ext_vector_type(16))) __bf16          v16bf;
typedef __attribute__((ext_vector_type(8)))  float           v8f;
typedef __attribute__((ext_vector_type(8)))  unsigned short  u16x8;
typedef __attribute__((ext_vector_type(4)))  int             v4i;

#define AS1 __attribute__((address_space(1)))
#define AS3 __attribute__((address_space(3)))

#define DEV static __device__ __forceinline__

// ---- bf16 helpers (bit-level) ---------------------------------------------
DEV unsigned short f32_to_bf16(float f) {
  unsigned int u = __float_as_uint(f);
  u += 0x7FFFu + ((u >> 16) & 1u);   // round-to-nearest-even
  return (unsigned short)(u >> 16);
}
DEV float bf16_to_f32(unsigned short s) {
  return __uint_as_float(((unsigned int)s) << 16);
}

// ---- CDNA5 async copy: global -> LDS, 16 bytes, tracked by ASYNCcnt --------
// Builtin signature (from clang diagnostic): (v4i AS1*, v4i AS3*, imm, imm).
#if __has_builtin(__builtin_amdgcn_global_load_async_to_lds_b128)
#define ASYNC_CP(gp, lp)                                                \
  __builtin_amdgcn_global_load_async_to_lds_b128(                       \
      (AS1 v4i*)(uintptr_t)(gp),                                        \
      (AS3 v4i*)(unsigned int)(uintptr_t)(lp), 0, 0)
#else
#define ASYNC_CP(gp, lp)                                                \
  asm volatile("global_load_async_to_lds_b128 %0, %1, off"              \
               :: "v"((unsigned int)(uintptr_t)(lp)),                   \
                  "v"((unsigned long long)(uintptr_t)(gp)) : "memory")
#endif

DEV void wait_async0() {
#if __has_builtin(__builtin_amdgcn_s_wait_asynccnt)
  __builtin_amdgcn_s_wait_asynccnt(0);
#else
  asm volatile("s_wait_asynccnt 0x0" ::: "memory");
#endif
}

// ---- WMMA fragment load from an LDS panel (row-major, 32 bf16 per row) -----
// 16-bit A/B layout per CDNA5 ISA 7.12.2: lane L holds row (L&15),
// elements 0..7 = K[kb..kb+7], elements 8..15 = K[16+kb..16+kb+7], kb=(L>>4)*8.
DEV v16bf lds_frag(const unsigned short* base, int lane) {
  const int r  = lane & 15;
  const int kb = (lane >> 4) << 3;
  const unsigned short* q = base + r * 32 + kb;
  v16bf f;
  u16x8* h = reinterpret_cast<u16x8*>(&f);
  h[0] = *reinterpret_cast<const u16x8*>(q);
  h[1] = *reinterpret_cast<const u16x8*>(q + 16);
  return f;
}

#define WMMA_BF16(a, b, c) \
  __builtin_amdgcn_wmma_f32_16x16x32_bf16(false, (a), false, (b), (short)0, (c), false, false)

// ---- GEMM: C[M,N] = act(A[M,K] * B[N,K]^T + bias) (+= C if accum) ----------
// 256 threads = 8 waves arranged 4(M) x 2(N); wave tile = 32x64 (2x4
// accumulators, 8 WMMA per K-step), block tile = 128 x 128.
// Requires M % 128 == 0, K % 32 == 0, N % 16 == 0 (partial N handled by
// source-row clamping + tile store guards). All call sites satisfy this.
__global__ __launch_bounds__(256)
void k_gemm_bf16_nt(const unsigned short* __restrict__ A, int lda,
                    const unsigned short* __restrict__ B, int ldb,
                    float* __restrict__ C, int ldc,
                    const float* __restrict__ bias,
                    int M, int N, int K, int act, int accum) {
  __shared__ __align__(16) unsigned short smA[2][128 * 32];
  __shared__ __align__(16) unsigned short smB[2][128 * 32];

  const int tid  = threadIdx.x;
  const int wave = tid >> 5;
  const int lane = tid & 31;
  const int wm   = wave & 3;   // 4 M-groups of 32 rows -> 128
  const int wn   = wave >> 2;  // 2 N-groups of 64 cols -> 128
  const int mblk = blockIdx.y * 128;
  const int nblk = blockIdx.x * 128;

  // Staging duty: each thread copies 2x16B of A-panel and 2x16B of B-panel
  // per K-step (128 rows x 32 cols bf16 = 8KB per panel).
  const int c0 = tid * 2;
  const int r0 = c0 >> 2,       q0 = (c0 & 3) << 3;        // row, col-offset
  const int r1 = (c0 + 1) >> 2, q1 = ((c0 + 1) & 3) << 3;
  const int bgr0 = min(nblk + r0, N - 1);                  // clamped B source
  const int bgr1 = min(nblk + r1, N - 1);

  v8f acc[2][4];
#pragma unroll
  for (int mt = 0; mt < 2; ++mt)
#pragma unroll
    for (int nt = 0; nt < 4; ++nt) acc[mt][nt] = v8f{};

  auto stage = [&](int k0, int p) {
    ASYNC_CP(A + (size_t)(mblk + r0) * lda + k0 + q0, &smA[p][r0 * 32 + q0]);
    ASYNC_CP(A + (size_t)(mblk + r1) * lda + k0 + q1, &smA[p][r1 * 32 + q1]);
    ASYNC_CP(B + (size_t)bgr0 * ldb + k0 + q0,        &smB[p][r0 * 32 + q0]);
    ASYNC_CP(B + (size_t)bgr1 * ldb + k0 + q1,        &smB[p][r1 * 32 + q1]);
  };

  const int nk = K >> 5;
  stage(0, 0);
  for (int kt = 0; kt < nk; ++kt) {
    const int p = kt & 1;
    wait_async0();       // my async copies into buffer p are done
    __syncthreads();     // => everyone's copies into p are done
    if (kt + 1 < nk) stage((kt + 1) << 5, p ^ 1);

    v16bf a0 = lds_frag(&smA[p][(wm * 32 + 0)  * 32], lane);
    v16bf a1 = lds_frag(&smA[p][(wm * 32 + 16) * 32], lane);
    v16bf b0 = lds_frag(&smB[p][(wn * 64 + 0)  * 32], lane);
    v16bf b1 = lds_frag(&smB[p][(wn * 64 + 16) * 32], lane);
    v16bf b2 = lds_frag(&smB[p][(wn * 64 + 32) * 32], lane);
    v16bf b3 = lds_frag(&smB[p][(wn * 64 + 48) * 32], lane);

    acc[0][0] = WMMA_BF16(a0, b0, acc[0][0]);
    acc[0][1] = WMMA_BF16(a0, b1, acc[0][1]);
    acc[0][2] = WMMA_BF16(a0, b2, acc[0][2]);
    acc[0][3] = WMMA_BF16(a0, b3, acc[0][3]);
    acc[1][0] = WMMA_BF16(a1, b0, acc[1][0]);
    acc[1][1] = WMMA_BF16(a1, b1, acc[1][1]);
    acc[1][2] = WMMA_BF16(a1, b2, acc[1][2]);
    acc[1][3] = WMMA_BF16(a1, b3, acc[1][3]);
  }

  // C/D layout: VGPR j -> row (j + (lane>=16 ? 8:0)), col = lane&15.
  const int col  = lane & 15;
  const int rsel = (lane >> 4) << 3;
#pragma unroll
  for (int mt = 0; mt < 2; ++mt) {
#pragma unroll
    for (int nt = 0; nt < 4; ++nt) {
      const int ntile = nblk + wn * 64 + nt * 16;
      if (ntile >= N) continue;                    // wave-uniform tile guard
      const int cc = ntile + col;
      const float bv = bias ? bias[cc] : 0.0f;
#pragma unroll
      for (int j = 0; j < 8; ++j) {
        const int rr = mblk + wm * 32 + mt * 16 + rsel + j;
        float v = acc[mt][nt][j] + bv;
        if (act == 1) v = (v > 20.0f) ? v : log1pf(__expf(v));  // softplus
        float* cp = C + (size_t)rr * ldc + cc;
        if (accum) v += *cp;
        *cp = v;
      }
    }
  }
}

// ---- prep: mask + bf16 convert, plus feature-axis flip for reverse dir -----
__global__ void k_prep(const float* __restrict__ x, const float* __restrict__ mask,
                       unsigned short* __restrict__ xf, unsigned short* __restrict__ xr,
                       int BL, int D) {
  const int idx = blockIdx.x * blockDim.x + threadIdx.x;
  if (idx >= BL * D) return;
  const int row = idx / D;
  const int d   = idx - row * D;
  const float v = x[idx] * mask[row];
  const unsigned short b = f32_to_bf16(v);
  xf[idx] = b;
  xr[(size_t)row * D + (D - 1 - d)] = b;  // jnp.flip(x, axis=-1)
}

// ---- generic f32 -> bf16 convert ------------------------------------------
__global__ void k_cvt_bf16(const float* __restrict__ in, unsigned short* __restrict__ out, int n) {
  const int idx = blockIdx.x * blockDim.x + threadIdx.x;
  if (idx < n) out[idx] = f32_to_bf16(in[idx]);
}

// ---- depthwise causal conv (width 4) + SiLU, output bf16 -------------------
__global__ void k_conv_silu(const float* __restrict__ u, const float* __restrict__ w,
                            const float* __restrict__ bias, unsigned short* __restrict__ out,
                            int Bn, int L, int C) {
  const int idx = blockIdx.x * blockDim.x + threadIdx.x;
  if (idx >= Bn * L * C) return;
  const int c  = idx % C;
  const int bl = idx / C;
  const int l  = bl % L;
  const int b  = bl / L;
  float acc = bias[c];
#pragma unroll
  for (int k = 0; k < 4; ++k) {
    const int lt = l + k - 3;
    if (lt >= 0) acc += u[((size_t)(b * L + lt)) * C + c] * w[c * 4 + k];
  }
  const float s = acc / (1.0f + __expf(-acc));  // SiLU
  out[idx] = f32_to_bf16(s);
}

// ---- selective scan: one thread per (b, d_inner) channel -------------------
__global__ void k_scan(const float* __restrict__ delta, const unsigned short* __restrict__ u,
                       const float* __restrict__ xdbl, const float* __restrict__ z,
                       const float* __restrict__ A_log, const float* __restrict__ Dp,
                       unsigned short* __restrict__ Y, int Bn, int L, int C) {
  const int gid = blockIdx.x * blockDim.x + threadIdx.x;
  if (gid >= Bn * C) return;
  const int b = gid / C;
  const int d = gid - b * C;

  float Arow[16];
#pragma unroll
  for (int s = 0; s < 16; ++s) Arow[s] = -__expf(A_log[(size_t)d * 16 + s]);
  const float Dd = Dp[d];

  float h[16];
#pragma unroll
  for (int s = 0; s < 16; ++s) h[s] = 0.0f;

  for (int l = 0; l < L; ++l) {
    const size_t base = (size_t)b * L + l;
    const float dlt = delta[base * C + d];
    const float uv  = bf16_to_f32(u[base * C + d]);
    const float* BC = xdbl + base * 96 + 64;   // Bs at [64..79], Cs at [80..95]
    const float dltu = dlt * uv;
    float y = 0.0f;
#pragma unroll
    for (int s = 0; s < 16; ++s) {
      const float dA = __expf(dlt * Arow[s]);
      h[s] = dA * h[s] + dltu * BC[s];
      y += h[s] * BC[16 + s];
    }
    const float zv = z[base * C + d];
    const float out = (y + uv * Dd) * (zv / (1.0f + __expf(-zv)));
    Y[base * C + d] = f32_to_bf16(out);
  }
}

// ---------------------------------------------------------------------------
// Host launcher
// ---------------------------------------------------------------------------
extern "C" void kernel_launch(void* const* d_in, const int* in_sizes, int n_in,
                              void* d_out, int out_size, void* d_ws, size_t ws_size,
                              hipStream_t stream) {
  (void)in_sizes; (void)n_in; (void)out_size; (void)ws_size;
  constexpr int Bn = 2, L = 1024, DM = 1024, DI = 2048;
  constexpr int BL = Bn * L;              // 2048 rows
  constexpr int XP = 96;                  // dt_rank + 2*d_state

  const float* x    = (const float*)d_in[0];
  const float* mask = (const float*)d_in[1];
  const float* w_in = (const float*)d_in[2];
  const float* w_out= (const float*)d_in[3];
  struct Dir { const float *conv_w, *conv_b, *xproj, *dtw, *dtb, *Alog, *Dp; };
  Dir dir[2] = {
    { (const float*)d_in[4],  (const float*)d_in[5],  (const float*)d_in[6],
      (const float*)d_in[7],  (const float*)d_in[8],  (const float*)d_in[9],
      (const float*)d_in[10] },
    { (const float*)d_in[11], (const float*)d_in[12], (const float*)d_in[13],
      (const float*)d_in[14], (const float*)d_in[15], (const float*)d_in[16],
      (const float*)d_in[17] },
  };

  // -- workspace carve-out (~91 MB) --
  char* wsp = (char*)d_ws;
  auto alloc = [&](size_t bytes) -> char* {
    char* p = wsp; wsp += (bytes + 255) & ~(size_t)255; return p;
  };
  unsigned short* Win_bf   = (unsigned short*)alloc((size_t)2 * DI * DM * 2);  // 4096x1024
  unsigned short* Wout_bf  = (unsigned short*)alloc((size_t)DM * DI * 2);      // 1024x2048
  unsigned short* Xf_bf    = (unsigned short*)alloc((size_t)BL * DM * 2);
  unsigned short* Xr_bf    = (unsigned short*)alloc((size_t)BL * DM * 2);
  unsigned short* xproj_bf = (unsigned short*)alloc((size_t)XP * DI * 2);
  unsigned short* dtw_bf   = (unsigned short*)alloc((size_t)DI * 64 * 2);
  float*          u_raw    = (float*)alloc((size_t)BL * DI * 4);
  float*          z_buf    = (float*)alloc((size_t)BL * DI * 4);
  unsigned short* uconv_bf = (unsigned short*)alloc((size_t)BL * DI * 2);
  float*          xdbl     = (float*)alloc((size_t)BL * XP * 4);
  unsigned short* xdbl_bf  = (unsigned short*)alloc((size_t)BL * XP * 2);
  float*          delta    = (float*)alloc((size_t)BL * DI * 4);
  unsigned short* Y_bf     = (unsigned short*)alloc((size_t)BL * DI * 2);

  auto cdiv = [](int a, int b) { return (a + b - 1) / b; };

  // one-time conversions + input prep
  k_cvt_bf16<<<cdiv(2 * DI * DM, 256), 256, 0, stream>>>(w_in, Win_bf, 2 * DI * DM);
  k_cvt_bf16<<<cdiv(DM * DI, 256), 256, 0, stream>>>(w_out, Wout_bf, DM * DI);
  k_prep<<<cdiv(BL * DM, 256), 256, 0, stream>>>(x, mask, Xf_bf, Xr_bf, BL, DM);

  for (int rdir = 0; rdir < 2; ++rdir) {
    const Dir& dd = dir[rdir];
    const unsigned short* Xb = rdir ? Xr_bf : Xf_bf;

    k_cvt_bf16<<<cdiv(XP * DI, 256), 256, 0, stream>>>(dd.xproj, xproj_bf, XP * DI);
    k_cvt_bf16<<<cdiv(DI * 64, 256), 256, 0, stream>>>(dd.dtw, dtw_bf, DI * 64);

    // u = X @ Win[0:2048]^T ; z = X @ Win[2048:4096]^T
    k_gemm_bf16_nt<<<dim3(cdiv(DI, 128), BL / 128), 256, 0, stream>>>(
        Xb, DM, Win_bf, DM, u_raw, DI, nullptr, BL, DI, DM, 0, 0);
    k_gemm_bf16_nt<<<dim3(cdiv(DI, 128), BL / 128), 256, 0, stream>>>(
        Xb, DM, Win_bf + (size_t)DI * DM, DM, z_buf, DI, nullptr, BL, DI, DM, 0, 0);

    // depthwise causal conv + SiLU
    k_conv_silu<<<cdiv(Bn * L * DI, 256), 256, 0, stream>>>(
        u_raw, dd.conv_w, dd.conv_b, uconv_bf, Bn, L, DI);

    // x_dbl = u_conv @ x_proj^T  (N = 96)
    k_gemm_bf16_nt<<<dim3(1, BL / 128), 256, 0, stream>>>(
        uconv_bf, DI, xproj_bf, DI, xdbl, XP, nullptr, BL, XP, DI, 0, 0);
    k_cvt_bf16<<<cdiv(BL * XP, 256), 256, 0, stream>>>(xdbl, xdbl_bf, BL * XP);

    // delta = softplus(dt @ dt_w^T + dt_b) ; dt = x_dbl[:, 0:64], lda = 96
    k_gemm_bf16_nt<<<dim3(cdiv(DI, 128), BL / 128), 256, 0, stream>>>(
        xdbl_bf, XP, dtw_bf, 64, delta, DI, dd.dtb, BL, DI, 64, 1, 0);

    // sequential selective scan -> Y (bf16)
    k_scan<<<cdiv(Bn * DI, 256), 256, 0, stream>>>(
        delta, uconv_bf, xdbl, z_buf, dd.Alog, dd.Dp, Y_bf, Bn, L, DI);

    // out (+)= Y @ Wout^T  (accumulate second direction)
    k_gemm_bf16_nt<<<dim3(cdiv(DM, 128), BL / 128), 256, 0, stream>>>(
        Y_bf, DI, Wout_bf, DI, (float*)d_out, DM, nullptr, BL, DM, DI, 0, rdir);
  }
}
